// STN_layer_62148176773700
// MI455X (gfx1250) — matrix operand order repaired and checked by reading, compiled, not run on hardware
//
#include <hip/hip_runtime.h>
#include <hip/hip_bf16.h>

typedef __attribute__((ext_vector_type(16))) _Float16 v16h;
typedef __attribute__((ext_vector_type(8)))  float    v8f;

#define BATCH 16
#define NPTS  4096
#define KNN   32
#define ROWS  256          // edge rows per workgroup (8 query points x 32 nbrs)

// ---------------------------------------------------------------------------
// WMMA fragment loaders (wave32, 16-bit, contiguous k = 16h + s, same order in A and B)
// A: 16x32 (MxK). lane&15 = M row; lanes>=16 get +16 K offset.
//    VGPR i -> K = 2i.
// B: 32x16 (KxN), stored here as rows of W (N-major, K contiguous).
//    lane&15 = N col; lanes>=16 get +16 K offset; VGPR j -> K = 2j.
// ---------------------------------------------------------------------------
__device__ __forceinline__ v16h load_a16(const _Float16* base, int ld, int m0,
                                         int k0, int lane) {
  const _Float16* p = base + (size_t)(m0 + (lane & 15)) * ld + k0 + ((lane >> 4) << 4);
  v16h f;
#pragma unroll
  for (int i = 0; i < 8; ++i) {
    int ko = 2 * i;
    f[2 * i]     = p[ko];
    f[2 * i + 1] = p[ko + 1];
  }
  return f;
}

__device__ __forceinline__ v16h load_b16(const _Float16* base, int ld, int k0,
                                         int n0, int lane) {
  const _Float16* p = base + (size_t)(n0 + (lane & 15)) * ld + k0 + ((lane >> 4) << 4);
  v16h f;
#pragma unroll
  for (int i = 0; i < 8; ++i) {
    f[2 * i]     = p[2 * i];
    f[2 * i + 1] = p[2 * i + 1];
  }
  return f;
}

__device__ __forceinline__ v8f wmma_f16(v16h a, v16h b, v8f c) {
  return __builtin_amdgcn_wmma_f32_16x16x32_f16(false, a, false, b, (short)0,
                                                c, false, false);
}

// ---------------------------------------------------------------------------
// Kernel 0: convert w2/w3 -> f16 workspace copies, zero the max-pool buffer.
// ---------------------------------------------------------------------------
__global__ __launch_bounds__(256) void prep_kernel(
    const float* __restrict__ w2, const float* __restrict__ w3,
    _Float16* __restrict__ w2h, _Float16* __restrict__ w3h,
    int* __restrict__ g) {
  int i = blockIdx.x * 256 + threadIdx.x;
  if (i < 128 * 64)   w2h[i] = (_Float16)w2[i];
  if (i < 1024 * 128) w3h[i] = (_Float16)w3[i];
  if (i < BATCH * 1024) g[i] = 0;
}

// ---------------------------------------------------------------------------
// Kernel 1: kNN. One thread per query point; candidates tiled through LDS.
// Rank by s = 2<xi,xj> - |xj|^2 (|xi|^2 is constant per row).
// ---------------------------------------------------------------------------
__global__ __launch_bounds__(256) void knn_kernel(const float* __restrict__ x,
                                                  int* __restrict__ idx) {
  __shared__ float xjs[256][3];
  __shared__ float sjs[256];
  int tid = threadIdx.x;
  int blk = blockIdx.x;
  int b = blk >> 4;                 // 16 blocks per batch (N/256)
  int n = ((blk & 15) << 8) + tid;
  const float* xb = x + (size_t)b * 3 * NPTS;
  float q0 = xb[n], q1 = xb[NPTS + n], q2 = xb[2 * NPTS + n];

  float bd[KNN];
  int   bi[KNN];
#pragma unroll
  for (int j = 0; j < KNN; ++j) { bd[j] = -3.0e38f; bi[j] = 0; }

  for (int m0 = 0; m0 < NPTS; m0 += 256) {
    float v0 = xb[m0 + tid];
    float v1 = xb[NPTS + m0 + tid];
    float v2 = xb[2 * NPTS + m0 + tid];
    __syncthreads();
    xjs[tid][0] = v0; xjs[tid][1] = v1; xjs[tid][2] = v2;
    sjs[tid] = v0 * v0 + v1 * v1 + v2 * v2;
    __syncthreads();
    for (int t = 0; t < 256; ++t) {
      float s = 2.0f * (q0 * xjs[t][0] + q1 * xjs[t][1] + q2 * xjs[t][2]) - sjs[t];
      if (s > bd[KNN - 1]) {
        int m = m0 + t;
#pragma unroll
        for (int j = 0; j < KNN; ++j) {
          if (s > bd[j]) {
            float td = bd[j]; int ti = bi[j];
            bd[j] = s; bi[j] = m;
            s = td; m = ti;
          }
        }
      }
    }
  }
  int* op = idx + ((size_t)b * NPTS + n) * KNN;
#pragma unroll
  for (int j = 0; j < KNN; ++j) op[j] = bi[j];
}

// ---------------------------------------------------------------------------
// Kernel 2: fused edge MLP 3->64->128->1024 + global max pool.
// One WG = 8 query points x 32 neighbors = 256 edge rows, 8 waves, ~100KB LDS
// (3 WGs / 320KB WGP). h2 GEMM: 256x128x64; h3 GEMM: 256x1024x128 with two
// N-tiles per pass so each A fragment feeds 8 WMMAs. Row-max folded per tile;
// ReLU realized by atomicMax against the zero-initialized pool buffer.
// ---------------------------------------------------------------------------
__global__ __launch_bounds__(256) void edge_mlp_kernel(
    const float* __restrict__ x, const int* __restrict__ idx,
    const float* __restrict__ w1, const float* __restrict__ b1,
    const _Float16* __restrict__ w2h, const float* __restrict__ b2,
    const _Float16* __restrict__ w3h, const float* __restrict__ b3,
    int* __restrict__ g) {
  __shared__ float    x0s[ROWS][3];
  __shared__ float    w1s[192];
  __shared__ float    b1s[64];
  __shared__ _Float16 h1[ROWS][64];
  __shared__ _Float16 h2[ROWS][128];

  int tid  = threadIdx.x;
  int lane = tid & 31;
  int wave = tid >> 5;
  int blk  = blockIdx.x;
  int b    = blk >> 9;              // 512 tiles per batch
  int n0q  = (blk & 511) << 3;      // 8 query points per WG

  // Warm this wave's 128-row slice of w3 (L2 -> L0), 256B per row.
  {
    const _Float16* wp = w3h + (size_t)(wave << 7) * 128;
#pragma unroll
    for (int r = 0; r < 4; ++r) {
      const _Float16* q = wp + (size_t)(r * 32 + lane) * 128;
      __builtin_prefetch(q, 0, 3);
      __builtin_prefetch(q + 64, 0, 3);
    }
  }

  // ---- stage w1/b1, gather edge features x_j - x_i ------------------------
  if (tid < 192) w1s[tid] = w1[tid];
  if (tid < 64)  b1s[tid] = b1[tid];
  {
    int e = tid;                    // one edge per thread
    int n = n0q + (e >> 5);
    int k = e & 31;
    int m = idx[((size_t)b * NPTS + n) * KNN + k];
    const float* xb = x + (size_t)b * 3 * NPTS;
#pragma unroll
    for (int c = 0; c < 3; ++c)
      x0s[e][c] = xb[c * NPTS + m] - xb[c * NPTS + n];
  }
  __syncthreads();

  // ---- h1 = relu(W1 x0 + b1): 3 -> 64 (VALU, output-major for LDS) --------
  {
    int o  = tid & 63;
    int eb = tid >> 6;              // 0..3
    float wa = w1s[o * 3 + 0], wb = w1s[o * 3 + 1], wc = w1s[o * 3 + 2];
    float bb = b1s[o];
#pragma unroll 4
    for (int j = 0; j < 64; ++j) {
      int e = (j << 2) + eb;
      float v = fmaf(wa, x0s[e][0], fmaf(wb, x0s[e][1], fmaf(wc, x0s[e][2], bb)));
      h1[e][o] = (_Float16)fmaxf(v, 0.0f);
    }
  }
  __syncthreads();

  // ---- h2 = relu(h1 W2^T + b2): 64 -> 128 (WMMA) --------------------------
  {
    int n0 = wave << 4;             // 16 output cols per wave
    v16h bf0 = load_b16(w2h, 64, 0,  n0, lane);
    v16h bf1 = load_b16(w2h, 64, 32, n0, lane);
    float bias = b2[n0 + (lane & 15)];
#pragma unroll 2
    for (int mt = 0; mt < ROWS / 16; ++mt) {
      v8f acc = {};
      v16h af0 = load_a16(&h1[0][0], 64, mt << 4, 0,  lane);
      v16h af1 = load_a16(&h1[0][0], 64, mt << 4, 32, lane);
      acc = wmma_f16(af0, bf0, acc);
      acc = wmma_f16(af1, bf1, acc);
      int mrow = (mt << 4) + ((lane >> 4) << 3);
      int col  = n0 + (lane & 15);
#pragma unroll
      for (int r = 0; r < 8; ++r)
        h2[mrow + r][col] = (_Float16)fmaxf(acc[r] + bias, 0.0f);
    }
  }
  __syncthreads();

  // ---- h3 = relu(h2 W3^T + b3): 128 -> 1024 (WMMA) + row-max -> atomicMax -
  // Wave covers 128 output cols; two 16-col N-tiles per pass so each A
  // fragment (loaded once from LDS) feeds 8 WMMAs.
  {
    int* gb = g + b * 1024;
#pragma unroll 1
    for (int np = 0; np < 4; ++np) {
      int nA = (wave << 7) + (np << 5);      // first 16-col tile
      int nB = nA + 16;                      // second 16-col tile
      v16h bA0 = load_b16(w3h, 128, 0,  nA, lane);
      v16h bA1 = load_b16(w3h, 128, 32, nA, lane);
      v16h bA2 = load_b16(w3h, 128, 64, nA, lane);
      v16h bA3 = load_b16(w3h, 128, 96, nA, lane);
      v16h bB0 = load_b16(w3h, 128, 0,  nB, lane);
      v16h bB1 = load_b16(w3h, 128, 32, nB, lane);
      v16h bB2 = load_b16(w3h, 128, 64, nB, lane);
      v16h bB3 = load_b16(w3h, 128, 96, nB, lane);
      float biasA = b3[nA + (lane & 15)];
      float biasB = b3[nB + (lane & 15)];
      float cmaxA = 0.0f, cmaxB = 0.0f;
#pragma unroll 2
      for (int mt = 0; mt < ROWS / 16; ++mt) {
        v16h a0 = load_a16(&h2[0][0], 128, mt << 4, 0,  lane);
        v16h a1 = load_a16(&h2[0][0], 128, mt << 4, 32, lane);
        v16h a2 = load_a16(&h2[0][0], 128, mt << 4, 64, lane);
        v16h a3 = load_a16(&h2[0][0], 128, mt << 4, 96, lane);
        v8f accA = {};
        accA = wmma_f16(a0, bA0, accA);
        accA = wmma_f16(a1, bA1, accA);
        accA = wmma_f16(a2, bA2, accA);
        accA = wmma_f16(a3, bA3, accA);
        v8f accB = {};
        accB = wmma_f16(a0, bB0, accB);
        accB = wmma_f16(a1, bB1, accB);
        accB = wmma_f16(a2, bB2, accB);
        accB = wmma_f16(a3, bB3, accB);
#pragma unroll
        for (int r = 0; r < 8; ++r) {
          cmaxA = fmaxf(cmaxA, accA[r] + biasA);
          cmaxB = fmaxf(cmaxB, accB[r] + biasB);
        }
      }
      // merge row halves (lanes l and l^16 hold the same column)
      cmaxA = fmaxf(cmaxA, __shfl_xor(cmaxA, 16, 32));
      cmaxB = fmaxf(cmaxB, __shfl_xor(cmaxB, 16, 32));
      if (lane < 16) {
        atomicMax(&gb[nA + lane], __float_as_int(cmaxA));  // values >= 0
        atomicMax(&gb[nB + lane], __float_as_int(cmaxB));
      }
    }
  }
}

// ---------------------------------------------------------------------------
// Kernel 3: FC head per batch: 1024 -> 512 -> 256 -> 9, + I(3x3).
// ---------------------------------------------------------------------------
__global__ __launch_bounds__(256) void head_kernel(
    const float* __restrict__ g,
    const float* __restrict__ fw1, const float* __restrict__ fb1,
    const float* __restrict__ fw2, const float* __restrict__ fb2,
    const float* __restrict__ fw3, const float* __restrict__ fb3,
    float* __restrict__ tmat) {
  __shared__ float gs[1024];
  __shared__ float f1[512];
  __shared__ float f2[256];
  int b = blockIdx.x, tid = threadIdx.x;
  const float* gb = g + b * 1024;
  for (int i = tid; i < 1024; i += 256) gs[i] = gb[i];
  __syncthreads();
  for (int o = tid; o < 512; o += 256) {
    float acc = fb1[o];
    const float* w = fw1 + (size_t)o * 1024;
    for (int i = 0; i < 1024; ++i) acc = fmaf(w[i], gs[i], acc);
    f1[o] = fmaxf(acc, 0.0f);
  }
  __syncthreads();
  if (tid < 256) {
    float acc = fb2[tid];
    const float* w = fw2 + (size_t)tid * 512;
    for (int i = 0; i < 512; ++i) acc = fmaf(w[i], f1[i], acc);
    f2[tid] = fmaxf(acc, 0.0f);
  }
  __syncthreads();
  if (tid < 9) {
    float acc = fb3[tid];
    const float* w = fw3 + (size_t)tid * 256;
    for (int i = 0; i < 256; ++i) acc = fmaf(w[i], f2[i], acc);
    if (tid == 0 || tid == 4 || tid == 8) acc += 1.0f;   // + eye(3)
    tmat[b * 9 + tid] = acc;
  }
}

// ---------------------------------------------------------------------------
// Kernel 4: out[b,d,n] = sum_c x[b,c,n] * t[b,c,d]
// ---------------------------------------------------------------------------
__global__ __launch_bounds__(256) void apply_kernel(
    const float* __restrict__ x, const float* __restrict__ tmat,
    float* __restrict__ out) {
  __shared__ float ts[9];
  int tid = threadIdx.x;
  int blk = blockIdx.x;
  int b = blk >> 4;
  int n = ((blk & 15) << 8) + tid;
  if (tid < 9) ts[tid] = tmat[b * 9 + tid];
  __syncthreads();
  const float* xb = x + (size_t)b * 3 * NPTS;
  float x0 = xb[n], x1 = xb[NPTS + n], x2 = xb[2 * NPTS + n];
  float* ob = out + (size_t)b * 3 * NPTS;
#pragma unroll
  for (int d = 0; d < 3; ++d)
    ob[d * NPTS + n] = x0 * ts[d] + x1 * ts[3 + d] + x2 * ts[6 + d];
}

// ---------------------------------------------------------------------------
extern "C" void kernel_launch(void* const* d_in, const int* in_sizes, int n_in,
                              void* d_out, int out_size, void* d_ws,
                              size_t ws_size, hipStream_t stream) {
  const float* x   = (const float*)d_in[0];
  const float* w1  = (const float*)d_in[1];
  const float* b1  = (const float*)d_in[2];
  const float* w2  = (const float*)d_in[3];
  const float* b2  = (const float*)d_in[4];
  const float* w3  = (const float*)d_in[5];
  const float* b3  = (const float*)d_in[6];
  const float* fw1 = (const float*)d_in[7];
  const float* fb1 = (const float*)d_in[8];
  const float* fw2 = (const float*)d_in[9];
  const float* fb2 = (const float*)d_in[10];
  const float* fw3 = (const float*)d_in[11];
  const float* fb3 = (const float*)d_in[12];

  char* ws = (char*)d_ws;
  size_t off = 0;
  int* idx = (int*)(ws + off);           off += (size_t)BATCH * NPTS * KNN * 4;
  _Float16* w2h = (_Float16*)(ws + off); off += 128 * 64 * 2;
  _Float16* w3h = (_Float16*)(ws + off); off += 1024 * 128 * 2;
  off = (off + 255) & ~(size_t)255;
  int* g = (int*)(ws + off);             off += BATCH * 1024 * 4;
  float* tmat = (float*)(ws + off);      off += BATCH * 9 * 4;

  prep_kernel<<<512, 256, 0, stream>>>(w2, w3, w2h, w3h, g);
  knn_kernel<<<BATCH * NPTS / 256, 256, 0, stream>>>(x, idx);
  edge_mlp_kernel<<<BATCH * (NPTS / 8), 256, 0, stream>>>(
      x, idx, w1, b1, w2h, b2, w3h, b3, g);
  head_kernel<<<BATCH, 256, 0, stream>>>((const float*)g, fw1, fb1, fw2, fb2,
                                         fw3, fb3, tmat);
  apply_kernel<<<BATCH * (NPTS / 256), 256, 0, stream>>>(x, tmat,
                                                         (float*)d_out);
}
